// SelectiveStateSpaceModel_10170482557535
// MI455X (gfx1250) — compile-verified, matrix operand order checked by reading, not run on hardware
//
#include <hip/hip_runtime.h>

// ---------------------------------------------------------------------------
// Selective SSM (gated GRU) for MI455X / gfx1250, wave32 + WMMA bf16 + async.
//
//  * softmax over a size-1 axis == 1.0  -> attention block is identity, dropped.
//  * One fused per-step GEMM: v=[x_t;h] (64x1536 bf16) times packed weights,
//    5 planes {r, z, inn, hn, gate}; inn only K<512, hn only K>=512 -> the
//    K loop is split into two branchless phases (4 WMMAs each).
//  * Weights 15.7 MB bf16: resident in 192 MB L2, re-streamed every step.
//  * x pre-converted to bf16 once; per-step A staging is a pure bf16 copy
//    done with global_load_async_to_lds_b128 (ASYNCcnt) into LDS.
//  * Persistent kernel, 32 WGs x 256 thr, grid barrier per step.
// ---------------------------------------------------------------------------

typedef __bf16 bf16;
typedef __attribute__((ext_vector_type(16))) __bf16 v16bf;
typedef __attribute__((ext_vector_type(8)))  __bf16 v8bf;
typedef __attribute__((ext_vector_type(8)))  float  v8f;

#define B_      64
#define S_      512
#define D_      512
#define H_      1024
#define K_      1536            // D + H
#define KT_     48              // K / 32
#define NT_     64              // H / 16
#define NPLANE  5               // r, z, inn, hn, gate
#define LDSTRIDE 1544           // 1536 + 8 bf16 pad (row starts bank-rotated)
#define NBLOCK  32
#define NTHREAD 256

// ---------------- weight packing: WMMA-B per-lane swizzle -------------------
__global__ void pack_weights(const float* __restrict__ Wg,
                             const float* __restrict__ W_ih,
                             const float* __restrict__ W_hh,
                             bf16* __restrict__ Wpack) {
  int idx = blockIdx.x * NTHREAD + threadIdx.x;
  const int total = NPLANE * KT_ * NT_ * 512;
  if (idx >= total) return;
  int within = idx & 511;
  int tile   = idx >> 9;
  int lane = within >> 4, e = within & 15;
  int nt = tile % NT_;
  int kt = (tile / NT_) % KT_;
  int p  = tile / (NT_ * KT_);
  int g  = lane >> 4;
  int kl = (e < 8) ? (g * 8 + e) : (16 + g * 8 + (e - 8));
  int k  = kt * 32 + kl;
  int j  = nt * 16 + (lane & 15);
  float w;
  if (p == 0)       w = (k < D_) ? W_ih[k * (3 * H_) + j]
                                 : W_hh[(k - D_) * (3 * H_) + j];
  else if (p == 1)  w = (k < D_) ? W_ih[k * (3 * H_) + H_ + j]
                                 : W_hh[(k - D_) * (3 * H_) + H_ + j];
  else if (p == 2)  w = (k < D_) ? W_ih[k * (3 * H_) + 2 * H_ + j] : 0.f;
  else if (p == 3)  w = (k < D_) ? 0.f : W_hh[(k - D_) * (3 * H_) + 2 * H_ + j];
  else              w = Wg[k * H_ + j];
  Wpack[idx] = (bf16)w;
}

// ----------------- x -> bf16 one-time conversion ----------------------------
__global__ void pack_x(const float* __restrict__ x, bf16* __restrict__ xbf) {
  size_t i = (size_t)blockIdx.x * NTHREAD + threadIdx.x;
  xbf[i] = (bf16)x[i];   // grid sized exactly to B_*S_*D_
}

// ------------- init: folded biases, zero h state, zero barrier --------------
__global__ void init_state(const float* __restrict__ bg,
                           const float* __restrict__ b_ih,
                           const float* __restrict__ b_hh,
                           float* __restrict__ bias,
                           float* __restrict__ h32,
                           bf16* __restrict__ hbf,
                           unsigned* __restrict__ cnt) {
  int i = blockIdx.x * NTHREAD + threadIdx.x;
  if (i < NPLANE * H_) {
    int p = i / H_, j = i % H_;
    float b;
    if (p == 0)      b = b_ih[j] + b_hh[j];
    else if (p == 1) b = b_ih[H_ + j] + b_hh[H_ + j];
    else if (p == 2) b = b_ih[2 * H_ + j];
    else if (p == 3) b = b_hh[2 * H_ + j];
    else             b = bg[j];
    bias[i] = b;
  }
  if (i < 2 * B_ * H_) { h32[i] = 0.f; hbf[i] = (bf16)0.f; }
  if (i == 0) *cnt = 0u;
}

__device__ __forceinline__ float sigmoidf_(float v) {
  return 1.f / (1.f + __expf(-v));
}

// CDNA5 async copy: 16 bytes per lane, global -> LDS, tracked by ASYNCcnt.
// ldsOff is the byte offset inside the workgroup's LDS allocation (dynamic
// LDS base == 0 here: no static __shared__ in this kernel).
__device__ __forceinline__ void async_copy_b128(unsigned ldsOff, const void* g) {
  unsigned long long ga = (unsigned long long)(uintptr_t)g;
  asm volatile("global_load_async_to_lds_b128 %0, %1, off"
               :: "v"(ldsOff), "v"(ga) : "memory");
}
__device__ __forceinline__ void wait_async0() {
  asm volatile("s_wait_asynccnt 0" ::: "memory");
}

// --------------------------- persistent recurrence --------------------------
__global__ void __launch_bounds__(NTHREAD, 1)
ssm_persistent(const bf16*  __restrict__ xbf,
               const bf16*  __restrict__ Wpack,
               const float* __restrict__ bias,
               float*       __restrict__ h32,   // [2][B_*H_] ping-pong
               bf16*        __restrict__ hbf,   // [2][B_*H_] ping-pong
               unsigned*    __restrict__ cnt,
               float*       __restrict__ out) {
  extern __shared__ bf16 Ab[];                  // [64][LDSTRIDE] bf16
  const int tid  = threadIdx.x;
  const int wave = tid >> 5;
  const int lane = tid & 31;
  const int mtile = wave & 3;                   // 4 M tiles (64 rows)
  const int nsub  = wave >> 2;                  // 2 N tiles per block
  const int nt    = blockIdx.x * 2 + nsub;      // 0..63
  const int g     = lane >> 4;
  const int col   = nt * 16 + (lane & 15);      // output h column

  const float b_r  = bias[0 * H_ + col];
  const float b_z  = bias[1 * H_ + col];
  const float b_n  = bias[2 * H_ + col];
  const float b_hn = bias[3 * H_ + col];
  const float b_g  = bias[4 * H_ + col];

  const size_t pstride = (size_t)KT_ * NT_ * 512;  // elements per plane
  const int arow = mtile * 16 + (lane & 15);
  const bf16* abase = Ab + arow * LDSTRIDE + g * 8;

  // Precomputed per-thread async-copy indices:
  //   x part: 4096 chunks of 8 bf16; chunk c -> row c>>6, col8 c&63
  //   h part: 8192 chunks of 8 bf16; chunk c -> row c>>7, col8 c&127
  for (int t = 0; t < S_; ++t) {
    const int cur = t & 1, nxt = cur ^ 1;
    const float* hcur  = h32 + cur * (B_ * H_);
    float*       hnext = h32 + nxt * (B_ * H_);
    const bf16*  hcbf  = hbf + cur * (B_ * H_);
    bf16*        hnbf  = hbf + nxt * (B_ * H_);
    const bf16*  xrow  = xbf + (size_t)t * D_;

    // ---- stage A = [x_t | h_{t-1}] into LDS via async copies ----
    for (int c = tid; c < (B_ * D_) / 8; c += NTHREAD) {
      int r = c >> 6, c8 = c & 63;
      async_copy_b128((unsigned)((r * LDSTRIDE + c8 * 8) * 2),
                      xrow + (size_t)r * (S_ * D_) + c8 * 8);
    }
    for (int c = tid; c < (B_ * H_) / 8; c += NTHREAD) {
      int r = c >> 7, c8 = c & 127;
      async_copy_b128((unsigned)((r * LDSTRIDE + D_ + c8 * 8) * 2),
                      hcbf + r * H_ + c8 * 8);
    }
    wait_async0();
    __syncthreads();

    // ---- K loop, two branchless phases, 4 independent WMMA chains each ----
    v8f accr = {}, accz = {}, accn = {}, acchn = {}, accg = {};
    const bf16* bb = Wpack + (size_t)nt * 512 + lane * 16;

#pragma unroll 2
    for (int kt = 0; kt < 16; ++kt) {            // x-rows: r,z,g + inn
      v8bf alo = *(const v8bf*)(abase + kt * 32);
      v8bf ahi = *(const v8bf*)(abase + kt * 32 + 16);
      v16bf a = __builtin_shufflevector(alo, ahi,
          0,1,2,3,4,5,6,7,8,9,10,11,12,13,14,15);
      const bf16* bk = bb + (size_t)kt * (NT_ * 512);
      v16bf br  = *(const v16bf*)(bk + 0 * pstride);
      v16bf bz  = *(const v16bf*)(bk + 1 * pstride);
      v16bf bn  = *(const v16bf*)(bk + 2 * pstride);
      v16bf bgt = *(const v16bf*)(bk + 4 * pstride);
      accr = __builtin_amdgcn_wmma_f32_16x16x32_bf16(false, a, false, br,
                                                     (short)0, accr, false, false);
      accz = __builtin_amdgcn_wmma_f32_16x16x32_bf16(false, a, false, bz,
                                                     (short)0, accz, false, false);
      accn = __builtin_amdgcn_wmma_f32_16x16x32_bf16(false, a, false, bn,
                                                     (short)0, accn, false, false);
      accg = __builtin_amdgcn_wmma_f32_16x16x32_bf16(false, a, false, bgt,
                                                     (short)0, accg, false, false);
    }
#pragma unroll 2
    for (int kt = 16; kt < KT_; ++kt) {          // h-rows: r,z,g + hn
      v8bf alo = *(const v8bf*)(abase + kt * 32);
      v8bf ahi = *(const v8bf*)(abase + kt * 32 + 16);
      v16bf a = __builtin_shufflevector(alo, ahi,
          0,1,2,3,4,5,6,7,8,9,10,11,12,13,14,15);
      const bf16* bk = bb + (size_t)kt * (NT_ * 512);
      v16bf br  = *(const v16bf*)(bk + 0 * pstride);
      v16bf bz  = *(const v16bf*)(bk + 1 * pstride);
      v16bf bh  = *(const v16bf*)(bk + 3 * pstride);
      v16bf bgt = *(const v16bf*)(bk + 4 * pstride);
      accr = __builtin_amdgcn_wmma_f32_16x16x32_bf16(false, a, false, br,
                                                     (short)0, accr, false, false);
      accz = __builtin_amdgcn_wmma_f32_16x16x32_bf16(false, a, false, bz,
                                                     (short)0, accz, false, false);
      acchn = __builtin_amdgcn_wmma_f32_16x16x32_bf16(false, a, false, bh,
                                                      (short)0, acchn, false, false);
      accg = __builtin_amdgcn_wmma_f32_16x16x32_bf16(false, a, false, bgt,
                                                     (short)0, accg, false, false);
    }

    // ---- elementwise GRU + selective gate (attention == identity) ----
    // C/D layout: lane holds N=col; VGPR v -> row M = v + 8*(lane>>4).
#pragma unroll
    for (int v = 0; v < 8; ++v) {
      int row = mtile * 16 + v + 8 * g;
      float hold = hcur[row * H_ + col];
      float r  = sigmoidf_(accr[v] + b_r);
      float z  = sigmoidf_(accz[v] + b_z);
      float n  = tanhf(accn[v] + b_n + r * (acchn[v] + b_hn));
      float gg = sigmoidf_(accg[v] + b_g);
      float hv = gg * ((1.f - z) * n + z * hold);
      hnext[row * H_ + col] = hv;
      hnbf[row * H_ + col]  = (bf16)hv;
      if (t == S_ - 1) out[row * H_ + col] = hv;
    }

    // ---- grid barrier (monotonic counter; zeroed per launch) ----
    __syncthreads();
    __threadfence();
    if (tid == 0) {
      atomicAdd(cnt, 1u);
      unsigned target = (unsigned)(t + 1) * NBLOCK;
      while (__hip_atomic_load(cnt, __ATOMIC_ACQUIRE,
                               __HIP_MEMORY_SCOPE_AGENT) < target) {
        __builtin_amdgcn_s_sleep(1);
      }
      __threadfence();
    }
    __syncthreads();
  }
}

// ----------------------------------------------------------------------------
extern "C" void kernel_launch(void* const* d_in, const int* in_sizes, int n_in,
                              void* d_out, int out_size, void* d_ws, size_t ws_size,
                              hipStream_t stream) {
  const float* x    = (const float*)d_in[0];
  const float* Wg   = (const float*)d_in[1];
  const float* bg   = (const float*)d_in[2];
  const float* W_ih = (const float*)d_in[3];
  const float* b_ih = (const float*)d_in[4];
  const float* W_hh = (const float*)d_in[5];
  const float* b_hh = (const float*)d_in[6];
  // d_in[7..10] (Wa, ba, Ws, bs) unused: softmax over a size-1 axis == 1.0.
  (void)in_sizes; (void)n_in; (void)out_size; (void)ws_size;

  char* ws = (char*)d_ws;
  bf16*     Wpack = (bf16*)(ws + 0);            // 15,728,640 B
  float*    bias  = (float*)(ws + 15728640);    //     20,480 B
  float*    h32   = (float*)(ws + 15749120);    //    524,288 B (2 buffers)
  bf16*     hbf   = (bf16*)(ws + 16273408);     //    262,144 B (2 buffers)
  unsigned* cnt   = (unsigned*)(ws + 16535552); //        256 B (padded)
  bf16*     xbf   = (bf16*)(ws + 16535808);     // 33,554,432 B
  // total ~50.1 MB of workspace

  const int packElems = NPLANE * KT_ * NT_ * 512;
  pack_weights<<<packElems / NTHREAD, NTHREAD, 0, stream>>>(Wg, W_ih, W_hh, Wpack);
  pack_x<<<(B_ * S_ * D_) / NTHREAD, NTHREAD, 0, stream>>>(x, xbf);
  init_state<<<(2 * B_ * H_) / NTHREAD, NTHREAD, 0, stream>>>(
      bg, b_ih, b_hh, bias, h32, hbf, cnt);

  size_t ldsbytes = (size_t)B_ * LDSTRIDE * sizeof(bf16);  // 197,632 B
  ssm_persistent<<<NBLOCK, NTHREAD, ldsbytes, stream>>>(
      xbf, Wpack, bias, h32, hbf, cnt, (float*)d_out);
}
